// CPTModel_86629490360411
// MI455X (gfx1250) — compile-verified, matrix operand order checked by reading, not run on hardware
//
#include <hip/hip_runtime.h>
#include <hip/hip_bf16.h>
#include <math.h>

// ---------------- model constants ----------------
#define B_   4
#define T_   512
#define C_   1024
#define H_   16
#define HD_  64
#define L_   12
#define R_   16
#define M_   (B_ * T_)          // 2048 tokens
#define SCALING_ 2.0f           // 32 / 16
#define EPS_ 1e-5f

// ---------------- WMMA fragment types ----------------
typedef __attribute__((ext_vector_type(16))) __bf16 v16bf;
typedef __attribute__((ext_vector_type(8)))  float  v8f;

union ABfrag {
  v16bf v;
  unsigned int d[8];
  unsigned short u[16];
};

// Native bf16 conversion (fptrunc -> hardware cvt, RNE) instead of bit-twiddling.
__device__ inline unsigned short f2bf(float f) {
  union { __bf16 h; unsigned short u; } c;
  c.h = (__bf16)f;
  return c.u;
}
__device__ inline unsigned pack2bf(float lo, float hi) {
  union { __bf16 h[2]; unsigned u; } c;
  c.h[0] = (__bf16)lo;
  c.h[1] = (__bf16)hi;
  return c.u;
}

// A/B 16x32 bf16 fragment from an LDS row-major tile (row = l&15,
// K-chunks (l>=16 ? 8 : 0) and +16; 2 bf16 per VGPR, low K first).
__device__ inline void frag_from_lds(const unsigned short* t, int ld, ABfrag& fr) {
  int l   = threadIdx.x & 31;
  int row = l & 15;
  int k0  = (l >> 4) << 3;
  uint4 a = *(const uint4*)(t + row * ld + k0);
  uint4 b = *(const uint4*)(t + row * ld + k0 + 16);
  fr.d[0] = a.x; fr.d[1] = a.y; fr.d[2] = a.z; fr.d[3] = a.w;
  fr.d[4] = b.x; fr.d[5] = b.y; fr.d[6] = b.z; fr.d[7] = b.w;
}

// Same fragment, loaded straight from a row-major fp32 matrix (convert to bf16).
__device__ inline void frag_from_gf32(const float* base, int ld, ABfrag& fr) {
  int l   = threadIdx.x & 31;
  int row = l & 15;
  int k0  = (l >> 4) << 3;
  const float4* p = (const float4*)(base + (size_t)row * ld + k0);
  float4 a = p[0], b = p[1];
  const float4* q = (const float4*)(base + (size_t)row * ld + k0 + 16);
  float4 c = q[0], d = q[1];
  fr.d[0] = pack2bf(a.x, a.y); fr.d[1] = pack2bf(a.z, a.w);
  fr.d[2] = pack2bf(b.x, b.y); fr.d[3] = pack2bf(b.z, b.w);
  fr.d[4] = pack2bf(c.x, c.y); fr.d[5] = pack2bf(c.z, c.w);
  fr.d[6] = pack2bf(d.x, d.y); fr.d[7] = pack2bf(d.z, d.w);
}

__device__ inline v8f wmma_bf16(const ABfrag& a, const ABfrag& b, v8f c) {
  return __builtin_amdgcn_wmma_f32_16x16x32_bf16(
      /*neg_a=*/false, a.v, /*neg_b=*/false, b.v,
      /*c_mod=*/(short)0, c, /*reuse_a=*/false, /*reuse_b=*/false);
}

// gfx1250 async global->LDS copy (ASYNCcnt-tracked, bypasses VGPRs).
typedef __attribute__((address_space(3))) unsigned short lds_us_t;
__device__ inline void async_load_b128(const void* gaddr, const unsigned short* lds_generic) {
  unsigned lofs = (unsigned)(unsigned long long)(lds_us_t*)lds_generic;
  asm volatile("global_load_async_to_lds_b128 %0, %1, off"
               :: "v"(lofs), "v"(gaddr) : "memory");
}
__device__ inline void wait_asynccnt0() {
#if __has_builtin(__builtin_amdgcn_s_wait_asynccnt)
  __builtin_amdgcn_s_wait_asynccnt(0);
#else
  asm volatile("s_wait_asynccnt 0" ::: "memory");
#endif
}

// ---------------- small utility kernels ----------------
__global__ void k_zero_u32(unsigned* p) { *p = 0u; }

__global__ void k_absmax(const float* __restrict__ w, long long n, unsigned* amax) {
  long long i = blockIdx.x * (long long)blockDim.x + threadIdx.x;
  float mx = 0.f;
  for (; i < n; i += (long long)gridDim.x * blockDim.x) mx = fmaxf(mx, fabsf(w[i]));
  #pragma unroll
  for (int o = 1; o < 32; o <<= 1) mx = fmaxf(mx, __shfl_xor(mx, o, 32));
  if ((threadIdx.x & 31) == 0) atomicMax(amax, __float_as_uint(mx));
}

// fq_sym fused with bf16 conversion: scale = absmax/127 + 1e-8
__global__ void k_quant_bf16(const float* __restrict__ w, long long n,
                             const unsigned* __restrict__ amax,
                             unsigned short* __restrict__ out) {
  float scale = __uint_as_float(*amax) * (1.0f / 127.0f) + 1e-8f;
  float inv   = 1.0f / scale;
  long long i = blockIdx.x * (long long)blockDim.x + threadIdx.x;
  for (; i < n; i += (long long)gridDim.x * blockDim.x) {
    float q = fminf(fmaxf(rintf(w[i] * inv), -128.f), 127.f) * scale;
    out[i] = f2bf(q);
  }
}

__global__ void k_embed(const int* __restrict__ ids, const float* __restrict__ wte,
                        const float* __restrict__ wpe, float* __restrict__ X) {
  int m = blockIdx.x;            // token
  int t = m % T_;
  int id = ids[m];
  int c = threadIdx.x * 4;
  float4 a = *(const float4*)(wte + (size_t)id * C_ + c);
  float4 p = *(const float4*)(wpe + (size_t)t * C_ + c);
  a.x += p.x; a.y += p.y; a.z += p.z; a.w += p.w;
  *(float4*)(X + (size_t)m * C_ + c) = a;
}

__global__ __launch_bounds__(256)
void k_layernorm(const float* __restrict__ X, const float* __restrict__ g,
                 const float* __restrict__ bt, float* __restrict__ Yo) {
  __shared__ float rs[8], rq[8];
  int m = blockIdx.x;
  int tid = threadIdx.x;
  const float* x = X + (size_t)m * C_;
  float4 v = *(const float4*)(x + tid * 4);
  float s = v.x + v.y + v.z + v.w;
  float q = v.x * v.x + v.y * v.y + v.z * v.z + v.w * v.w;
  #pragma unroll
  for (int o = 1; o < 32; o <<= 1) { s += __shfl_xor(s, o, 32); q += __shfl_xor(q, o, 32); }
  if ((tid & 31) == 0) { rs[tid >> 5] = s; rq[tid >> 5] = q; }
  __syncthreads();
  float ts = 0.f, tq = 0.f;
  #pragma unroll
  for (int i = 0; i < 8; ++i) { ts += rs[i]; tq += rq[i]; }
  float mean = ts * (1.0f / C_);
  float var  = tq * (1.0f / C_) - mean * mean;
  float inv  = rsqrtf(var + EPS_);
  float4 gg = *(const float4*)(g + tid * 4);
  float4 bb = *(const float4*)(bt + tid * 4);
  float4 o4;
  o4.x = (v.x - mean) * inv * gg.x + bb.x;
  o4.y = (v.y - mean) * inv * gg.y + bb.y;
  o4.z = (v.z - mean) * inv * gg.z + bb.z;
  o4.w = (v.w - mean) * inv * gg.w + bb.w;
  *(float4*)(Yo + (size_t)m * C_ + tid * 4) = o4;
}

__global__ void k_gelu(float* __restrict__ x, long long n) {
  long long i = blockIdx.x * (long long)blockDim.x + threadIdx.x;
  if (i < n) {
    float v = x[i];
    x[i] = 0.5f * v * (1.0f + erff(v * 0.70710678118654752f));
  }
}

__global__ void k_resid(float* __restrict__ x, const float* __restrict__ y, int n4) {
  int i = blockIdx.x * blockDim.x + threadIdx.x;
  if (i < n4) {
    float4 a = ((const float4*)x)[i];
    float4 b = ((const float4*)y)[i];
    a.x += b.x; a.y += b.y; a.z += b.z; a.w += b.w;
    ((float4*)x)[i] = a;
  }
}

// split qkv [M,3C] -> Q,K,V [B*H, T, 64]; Q pre-scaled by 1/sqrt(64);
// K,V pass through fq_asym (learnable scale/zp, qmax=255).
__global__ void k_split_qkv(const float* __restrict__ qkv,
                            const float* __restrict__ scale_p,
                            const float* __restrict__ zp_p,
                            float* __restrict__ Qh, float* __restrict__ Kh,
                            float* __restrict__ Vh) {
  float scale = *scale_p, zp = *zp_p;
  int idx = blockIdx.x * blockDim.x + threadIdx.x;      // over M_*C_
  if (idx >= M_ * C_) return;
  int m = idx / C_, c = idx % C_;
  int b = m / T_, t = m % T_;
  int h = c / HD_, d = c % HD_;
  size_t dst = (((size_t)(b * H_ + h)) * T_ + t) * HD_ + d;
  const float* row = qkv + (size_t)m * 3 * C_;
  Qh[dst] = row[c] * 0.125f;
  float kx = row[C_ + c];
  float qv = fminf(fmaxf(rintf(kx / scale) + zp, 0.f), 255.f);
  Kh[dst]  = (qv - zp) * scale;
  float vx = row[2 * C_ + c];
  qv = fminf(fmaxf(rintf(vx / scale) + zp, 0.f), 255.f);
  Vh[dst]  = (qv - zp) * scale;
}

// ---------------- LoRA first stage: t1[M,16] = X @ A^T ----------------
__global__ __launch_bounds__(32)
void k_lora_xa(const float* __restrict__ X, const float* __restrict__ LA,
               float* __restrict__ T1, int Kdim) {
  int mBase = blockIdx.x * 16;
  int lane = threadIdx.x & 31;
  int nlo = lane & 15, hsel = lane >> 4;
  v8f acc = (v8f){0, 0, 0, 0, 0, 0, 0, 0};
  for (int kk = 0; kk < Kdim; kk += 32) {
    ABfrag af, bfr;
    frag_from_gf32(X + (size_t)mBase * Kdim + kk, Kdim, af);
    frag_from_gf32(LA + kk, Kdim, bfr);   // rows of A are the N index
    acc = wmma_bf16(af, bfr, acc);
  }
  #pragma unroll
  for (int r = 0; r < 8; ++r) {
    int m = mBase + r + hsel * 8;
    T1[(size_t)m * R_ + nlo] = acc[r];
  }
}

// ---------------- main GEMM: Y = X @ Wq^T + bias + SCALING*(t1 @ loraB^T) ----------------
#define BM 128
#define BN 128
#define BK 32
#define LDT 40   // LDS row stride in ushorts (80B, 16B-aligned chunks)

__global__ __launch_bounds__(256)
void k_gemm_lora(const float* __restrict__ X, const unsigned short* __restrict__ Wq,
                 const float* __restrict__ bias, const float* __restrict__ t1,
                 const float* __restrict__ loraB, float scaling,
                 float* __restrict__ Y, int Ndim, int Kdim) {
  __shared__ __align__(16) unsigned short sA[BM * LDT];
  __shared__ __align__(16) unsigned short sB[BN * LDT];
  int mBase = blockIdx.y * BM, nBase = blockIdx.x * BN;
  int tid = threadIdx.x;
  int wid = tid >> 5;
  int lane = tid & 31;
  int nlo = lane & 15, hsel = lane >> 4;

  v8f acc[8];
  #pragma unroll
  for (int i = 0; i < 8; ++i) acc[i] = (v8f){0, 0, 0, 0, 0, 0, 0, 0};

  for (int kk = 0; kk < Kdim; kk += BK) {
    // B tile (bf16 weights): async global->LDS, no VGPR bounce. 512 x b128, 2/lane.
    #pragma unroll
    for (int i = 0; i < 2; ++i) {
      int f = tid + i * 256;
      int r = f >> 2;
      int c = (f & 3) << 3;
      async_load_b128(Wq + (size_t)(nBase + r) * Kdim + kk + c, sB + r * LDT + c);
    }
    // A tile (fp32 -> bf16 in registers): 128x32 = 1024 float4, 4 per thread
    #pragma unroll
    for (int i = 0; i < 4; ++i) {
      int f = tid + i * 256;
      int r = f >> 3;
      int c = (f & 7) << 2;
      const float* src = X + (size_t)(mBase + r) * Kdim + kk + c;
      if (kk + BK < Kdim) __builtin_prefetch(src + BK, 0, 1);
      float4 v = *(const float4*)src;
      *(uint2*)(sA + r * LDT + c) = make_uint2(pack2bf(v.x, v.y), pack2bf(v.z, v.w));
    }
    wait_asynccnt0();
    __syncthreads();
    ABfrag af;
    frag_from_lds(sA + (wid * 16) * LDT, LDT, af);
    #pragma unroll
    for (int nt = 0; nt < 8; ++nt) {
      ABfrag bfr;
      frag_from_lds(sB + (nt * 16) * LDT, LDT, bfr);
      acc[nt] = wmma_bf16(af, bfr, acc[nt]);
    }
    __syncthreads();   // DScnt drained here -> safe to overwrite tiles next iter
  }

  // epilogue: bias + LoRA second stage
  #pragma unroll
  for (int nt = 0; nt < 8; ++nt) {
    int n = nBase + nt * 16 + nlo;
    float bsv = bias[n];
    float lb[R_];
    #pragma unroll
    for (int j = 0; j < R_; ++j) lb[j] = loraB[(size_t)n * R_ + j];
    #pragma unroll
    for (int r = 0; r < 8; ++r) {
      int m = mBase + wid * 16 + r + hsel * 8;
      const float* t1r = t1 + (size_t)m * R_;
      float s = 0.f;
      #pragma unroll
      for (int j = 0; j < R_; ++j) s += t1r[j] * lb[j];
      Y[(size_t)m * Ndim + n] = acc[nt][r] + bsv + scaling * s;
    }
  }
}

// ---------------- flash attention: 1 wave per (b,h, 16-query tile) ----------------
#define LDP 40
__global__ __launch_bounds__(32)
void k_attn(const float* __restrict__ Qh, const float* __restrict__ Kh,
            const float* __restrict__ Vh, float* __restrict__ Out) {
  __shared__ __align__(16) unsigned short sP[16 * LDP];
  __shared__ __align__(16) unsigned short sVT[64 * LDP];
  const int nqt = T_ / 16;
  int bh = blockIdx.x / nqt;
  int qt = blockIdx.x % nqt;
  int b = bh / H_, h = bh % H_;
  const float* Qb = Qh + (size_t)bh * T_ * HD_;
  const float* Kb = Kh + (size_t)bh * T_ * HD_;
  const float* Vb = Vh + (size_t)bh * T_ * HD_;
  int lane = threadIdx.x & 31;
  int nlo = lane & 15, hsel = lane >> 4;

  ABfrag q0, q1;
  frag_from_gf32(Qb + (size_t)qt * 16 * HD_, HD_, q0);        // k 0..31
  frag_from_gf32(Qb + (size_t)qt * 16 * HD_ + 32, HD_, q1);   // k 32..63

  v8f o[4];
  #pragma unroll
  for (int i = 0; i < 4; ++i) o[i] = (v8f){0, 0, 0, 0, 0, 0, 0, 0};
  float m8[8], l8[8];
  #pragma unroll
  for (int r = 0; r < 8; ++r) { m8[r] = -1e30f; l8[r] = 0.f; }

  int qmx = qt * 16 + 15;
  int kcmax = qmx >> 5;
  for (int kc = 0; kc <= kcmax; ++kc) {
    v8f s0 = (v8f){0, 0, 0, 0, 0, 0, 0, 0};
    v8f s1 = (v8f){0, 0, 0, 0, 0, 0, 0, 0};
    const float* Kc = Kb + (size_t)kc * 32 * HD_;
    ABfrag kf;
    frag_from_gf32(Kc, HD_, kf);                 s0 = wmma_bf16(q0, kf, s0);
    frag_from_gf32(Kc + 32, HD_, kf);            s0 = wmma_bf16(q1, kf, s0);
    frag_from_gf32(Kc + 16 * HD_, HD_, kf);      s1 = wmma_bf16(q0, kf, s1);
    frag_from_gf32(Kc + 16 * HD_ + 32, HD_, kf); s1 = wmma_bf16(q1, kf, s1);

    // causal mask + online softmax (row = r + 8*hsel spread over 16 lanes)
    #pragma unroll
    for (int r = 0; r < 8; ++r) {
      int qrow = qt * 16 + r + hsel * 8;
      if (kc * 32 + nlo > qrow)      s0[r] = -1e30f;
      if (kc * 32 + 16 + nlo > qrow) s1[r] = -1e30f;
      float mx = fmaxf(s0[r], s1[r]);
      #pragma unroll
      for (int off = 1; off < 16; off <<= 1) mx = fmaxf(mx, __shfl_xor(mx, off, 32));
      float mn = fmaxf(m8[r], mx);
      float corr = __expf(m8[r] - mn);
      float p0 = __expf(s0[r] - mn);
      float p1 = __expf(s1[r] - mn);
      float rsum = p0 + p1;
      #pragma unroll
      for (int off = 1; off < 16; off <<= 1) rsum += __shfl_xor(rsum, off, 32);
      l8[r] = l8[r] * corr + rsum;
      m8[r] = mn;
      #pragma unroll
      for (int vt = 0; vt < 4; ++vt) o[vt][r] *= corr;
      int mrow = r + hsel * 8;
      sP[mrow * LDP + nlo]      = f2bf(p0);
      sP[mrow * LDP + 16 + nlo] = f2bf(p1);
    }

    // stage V chunk transposed (V^T: rows = d, cols = local k); lane owns k = lane
    const float* Vr = Vb + (size_t)(kc * 32 + lane) * HD_;
    #pragma unroll
    for (int dd = 0; dd < 64; dd += 4) {
      float4 v = *(const float4*)(Vr + dd);
      sVT[(dd + 0) * LDP + lane] = f2bf(v.x);
      sVT[(dd + 1) * LDP + lane] = f2bf(v.y);
      sVT[(dd + 2) * LDP + lane] = f2bf(v.z);
      sVT[(dd + 3) * LDP + lane] = f2bf(v.w);
    }
    __syncthreads();
    ABfrag pf;
    frag_from_lds(sP, LDP, pf);
    #pragma unroll
    for (int vt = 0; vt < 4; ++vt) {
      ABfrag vf;
      frag_from_lds(sVT + (vt * 16) * LDP, LDP, vf);
      o[vt] = wmma_bf16(pf, vf, o[vt]);
    }
    __syncthreads();
  }

  #pragma unroll
  for (int r = 0; r < 8; ++r) {
    int t = qt * 16 + r + hsel * 8;
    float inv = 1.0f / l8[r];
    #pragma unroll
    for (int vt = 0; vt < 4; ++vt) {
      int d = vt * 16 + nlo;
      Out[((size_t)(b * T_ + t)) * C_ + h * HD_ + d] = o[vt][r] * inv;
    }
  }
}

// ---------------- host orchestration ----------------
extern "C" void kernel_launch(void* const* d_in, const int* in_sizes, int n_in,
                              void* d_out, int out_size, void* d_ws, size_t ws_size,
                              hipStream_t stream) {
  (void)in_sizes; (void)n_in; (void)out_size; (void)ws_size;
  const int*   ids = (const int*)d_in[0];
  const float* wte = (const float*)d_in[1];
  const float* wpe = (const float*)d_in[2];
  auto LP = [&](int layer, int off) { return (const float*)d_in[3 + layer * 22 + off]; };
  const float* lnf_g = (const float*)d_in[3 + L_ * 22 + 0];
  const float* lnf_b = (const float*)d_in[3 + L_ * 22 + 1];

  char* ws = (char*)d_ws;
  size_t off = 0;
  auto carve = [&](size_t bytes) {
    void* p = ws + off;
    off = (off + bytes + 255) & ~(size_t)255;
    return p;
  };
  float* x    = (float*)carve(sizeof(float) * M_ * C_);
  float* hbuf = (float*)carve(sizeof(float) * M_ * C_);
  float* qkv  = (float*)carve(sizeof(float) * M_ * 3 * C_);
  float* Qhb  = (float*)carve(sizeof(float) * M_ * C_);
  float* Khb  = (float*)carve(sizeof(float) * M_ * C_);
  float* Vhb  = (float*)carve(sizeof(float) * M_ * C_);
  float* att  = (float*)carve(sizeof(float) * M_ * C_);
  float* f1   = (float*)carve(sizeof(float) * M_ * 4 * C_);
  float* t1   = (float*)carve(sizeof(float) * M_ * R_);
  unsigned short* wq = (unsigned short*)carve(sizeof(unsigned short) * 4 * C_ * C_);
  unsigned* amax = (unsigned*)carve(256);

  auto quantize = [&](const float* w, long long n) {
    k_zero_u32<<<1, 1, 0, stream>>>(amax);
    k_absmax<<<512, 256, 0, stream>>>(w, n, amax);
    k_quant_bf16<<<1024, 256, 0, stream>>>(w, n, amax, wq);
  };

  k_embed<<<M_, 256, 0, stream>>>(ids, wte, wpe, x);

  for (int l = 0; l < L_; ++l) {
    // ---- attention sub-block ----
    k_layernorm<<<M_, 256, 0, stream>>>(x, LP(l, 0), LP(l, 1), hbuf);

    quantize(LP(l, 2), (long long)3 * C_ * C_);
    k_lora_xa<<<M_ / 16, 32, 0, stream>>>(hbuf, LP(l, 4), t1, C_);
    k_gemm_lora<<<dim3(3 * C_ / BN, M_ / BM), 256, 0, stream>>>(
        hbuf, wq, LP(l, 3), t1, LP(l, 5), SCALING_, qkv, 3 * C_, C_);

    k_split_qkv<<<(M_ * C_) / 256, 256, 0, stream>>>(qkv, LP(l, 10), LP(l, 11),
                                                     Qhb, Khb, Vhb);
    k_attn<<<B_ * H_ * (T_ / 16), 32, 0, stream>>>(Qhb, Khb, Vhb, att);

    quantize(LP(l, 6), (long long)C_ * C_);
    k_lora_xa<<<M_ / 16, 32, 0, stream>>>(att, LP(l, 8), t1, C_);
    k_gemm_lora<<<dim3(C_ / BN, M_ / BM), 256, 0, stream>>>(
        att, wq, LP(l, 7), t1, LP(l, 9), SCALING_, hbuf, C_, C_);
    k_resid<<<(M_ * C_ / 4 + 255) / 256, 256, 0, stream>>>(x, hbuf, M_ * C_ / 4);

    // ---- MLP sub-block ----
    k_layernorm<<<M_, 256, 0, stream>>>(x, LP(l, 12), LP(l, 13), hbuf);

    quantize(LP(l, 14), (long long)4 * C_ * C_);
    k_lora_xa<<<M_ / 16, 32, 0, stream>>>(hbuf, LP(l, 16), t1, C_);
    k_gemm_lora<<<dim3(4 * C_ / BN, M_ / BM), 256, 0, stream>>>(
        hbuf, wq, LP(l, 15), t1, LP(l, 17), SCALING_, f1, 4 * C_, C_);

    k_gelu<<<(int)(((long long)M_ * 4 * C_ + 255) / 256), 256, 0, stream>>>(
        f1, (long long)M_ * 4 * C_);

    quantize(LP(l, 18), (long long)C_ * 4 * C_);
    k_lora_xa<<<M_ / 16, 32, 0, stream>>>(f1, LP(l, 20), t1, 4 * C_);
    k_gemm_lora<<<dim3(C_ / BN, M_ / BM), 256, 0, stream>>>(
        f1, wq, LP(l, 19), t1, LP(l, 21), SCALING_, hbuf, C_, 4 * C_);
    k_resid<<<(M_ * C_ / 4 + 255) / 256, 256, 0, stream>>>(x, hbuf, M_ * C_ / 4);
  }

  k_layernorm<<<M_, 256, 0, stream>>>(x, lnf_g, lnf_b, (float*)d_out);
}
